// MHAAttention_67662914781407
// MI455X (gfx1250) — matrix operand order repaired and checked by reading, not compiled, not run on hardware
//
#include <hip/hip_runtime.h>
#include <math.h>

// ---------------------------------------------------------------------------
// MHA + RoPE for MI455X (gfx1250), wave32.
//  - All matmuls on v_wmma_f32_16x16x32_f16 (f16 in, fp32 accumulate).
//  - Projection / output GEMMs: double-buffered GLOBAL_LOAD_ASYNC_TO_LDS_B128
//    staging (ASYNCcnt); workgroup tile 128x64, wave tile 16x64 (c[4]).
//    Single barrier per K-step: wait -> barrier -> prefetch -> compute.
//    All 5 fragments preloaded per K-step so ds_load latency overlaps WMMAs.
//  - Attention: flash-style, online softmax in registers, P transposed through
//    per-wave LDS with s_wait_dscnt.
// Workspace (~118 MB assumed): Xh, Wq/Wk/Wv/Wo (f16), Q/K [b,h,t,d],
// V [b,h,d,t] (transposed so PV B-fragments are contiguous), CTX.
// ---------------------------------------------------------------------------

typedef _Float16 f16;
typedef __attribute__((ext_vector_type(8)))  f16   v8h;
typedef __attribute__((ext_vector_type(16))) f16   v16h;
typedef __attribute__((ext_vector_type(8)))  float v8f;

static constexpr int BATCH = 2, SEQ = 2048, DM = 2048, NH = 16, DH = 128;
static constexpr int ROWS  = BATCH * SEQ;             // 4096
static constexpr int LP    = 40;                      // LDS panel pitch (f16), pad vs 32

#define DEV __device__ __forceinline__

DEV v16h cat16(v8h lo, v8h hi) {
  return __builtin_shufflevector(lo, hi, 0,1,2,3,4,5,6,7,8,9,10,11,12,13,14,15);
}

DEV v8f wmma_f16(v16h a, v16h b, v8f c) {
  return __builtin_amdgcn_wmma_f32_16x16x32_f16(false, a, false, b,
                                                (short)0, c, false, false);
}

// flat LDS pointer -> raw LDS byte offset (flat LDS = {SHARED_BASE, off[31:0]})
DEV unsigned lds_off(const void* p) { return (unsigned)(size_t)p; }

// async 16B global -> LDS copy (ASYNCcnt). No "memory" clobber here: the
// s_wait_asynccnt asm carries it, so the compiler keeps addresses lean.
DEV void async_cp16(unsigned lds_byte_off, const f16* g) {
  asm volatile("global_load_async_to_lds_b128 %0, %1, off"
               :: "v"(lds_byte_off), "v"((unsigned long long)(size_t)g));
}

// A-fragment 16x32 f16 (row-major source, pitch in f16 elements):
// lane: row = row0 + lane%16; elems0..7 @K = k+8*(lane/16), elems8..15 @K+16.
DEV v16h load_a_frag(const f16* __restrict__ base, int pitch, int row0, int k, int lane) {
  const f16* p = base + (size_t)(row0 + (lane & 15)) * pitch + k + 8 * (lane >> 4);
  return cat16(*(const v8h*)(p), *(const v8h*)(p + 16));
}

// B-fragment 32x16: B[k][n] = src[col0+n][k..]; lane: n = lane%16,
// elems0..7 @K = k+8*(lane/16), elems8..15 @K+16.
DEV v16h load_b_frag(const f16* __restrict__ base, int pitch, int col0, int k, int lane) {
  const f16* p = base + (size_t)(col0 + (lane & 15)) * pitch + k + 8 * (lane >> 4);
  return cat16(*(const v8h*)(p), *(const v8h*)(p + 16));
}

// ---------------------------------------------------------------------------
// fp32 -> f16 conversion
// ---------------------------------------------------------------------------
__global__ void cvt_f32_f16_kernel(const float* __restrict__ src,
                                   f16* __restrict__ dst, int n) {
  int i = blockIdx.x * blockDim.x + threadIdx.x;
  int stride = gridDim.x * blockDim.x;
  for (; i < n; i += stride) dst[i] = (f16)src[i];
}

// ---------------------------------------------------------------------------
// Double-buffered async-LDS GEMM mainloop. Workgroup = 8 waves = 128x64 tile;
// wave wv owns rows 16*wv..16*wv+15 and all 64 columns (c[4]).
// CS (col-split): B rows r<32 -> e0+r, r>=32 -> e0+32+r  (RoPE-friendly
// quarters {g, g+2} of a head); CS=false: contiguous e0+r.
// ---------------------------------------------------------------------------
struct GemmSmem {
  __align__(16) f16 A[2][128 * LP];   // 10 KB per buffer
  __align__(16) f16 B[2][64 * LP];    //  5 KB per buffer
};

template <bool CS>
DEV void gemm_mainloop(GemmSmem& s, const f16* __restrict__ X, const f16* __restrict__ W,
                       int r0, int e0, int wv, int tid, int lane, v8f c[4]) {
  // staging addresses hoisted out of the loop (3 async b128 per thread/step)
  const int crow = tid >> 2, cc = tid & 3;                  // 16B chunks, 4 per 64B row
  const int be   = e0 + crow + ((CS && crow >= 32) ? 32 : 0);
  const f16* gA  = X + (size_t)(r0 + crow) * DM + cc * 8;   // second chunk at +64 rows
  const f16* gB  = W + (size_t)be * DM + cc * 8;
  const unsigned lA   = lds_off(&s.A[0][crow * LP + cc * 8]);
  const unsigned lB   = lds_off(&s.B[0][crow * LP + cc * 8]);
  const unsigned bufA = sizeof(s.A[0]), bufB = sizeof(s.B[0]);
  const unsigned lA2  = 64u * LP * sizeof(f16);             // rows 64..127 of A panel

  async_cp16(lA,       gA);                                 // prologue: buf 0, kk = 0
  async_cp16(lA + lA2, gA + (size_t)64 * DM);
  async_cp16(lB,       gB);

#pragma unroll 1        // keep one body: accumulators stay register-tied
  for (int kk = 0, it = 0; kk < DM; kk += 32, ++it) {
    const int buf = it & 1;
    asm volatile("s_wait_asynccnt 0x0" ::: "memory");       // own panel loads done
    __syncthreads();    // all panels visible; also: everyone finished prev compute
    if (kk + 32 < DM) { // prefetch into buf^1 (fully drained by the barrier above)
      const unsigned nb = (unsigned)(buf ^ 1);
      async_cp16(lA + nb * bufA,       gA + kk + 32);
      async_cp16(lA + nb * bufA + lA2, gA + kk + 32 + (size_t)64 * DM);
      async_cp16(lB + nb * bufB,       gB + kk + 32);
    }
    const f16* Ab = &s.A[buf][0];
    const f16* Bb = &s.B[buf][0];
    // preload all fragments (10 ds_load_b128 in flight), then 4 WMMAs with
    // staggered s_wait_dscnt so LDS latency overlaps the matrix pipe
    v16h a  = load_a_frag(Ab, LP, 16 * wv, 0, lane);
    v16h b0 = load_b_frag(Bb, LP,  0, 0, lane);
    v16h b1 = load_b_frag(Bb, LP, 16, 0, lane);
    v16h b2 = load_b_frag(Bb, LP, 32, 0, lane);
    v16h b3 = load_b_frag(Bb, LP, 48, 0, lane);
    c[0] = wmma_f16(a, b0, c[0]);
    c[1] = wmma_f16(a, b1, c[1]);
    c[2] = wmma_f16(a, b2, c[2]);
    c[3] = wmma_f16(a, b3, c[3]);
  }
}

// ---------------------------------------------------------------------------
// QKV projection. Workgroup: 128 rows x 64 cols = head quarters {g, g+2}, so
// RoPE pairs (dp, dp+64) sit in tiles (j, j+2), same lane, same VGPR.
// mode: 0=Q (RoPE + fold log2e/sqrt(128), [b,h,t,d]) 1=K (RoPE, [b,h,t,d])
//       2=V (no RoPE, transposed [b,h,d,t])
// ---------------------------------------------------------------------------
__global__ void proj_kernel(const f16* __restrict__ X, const f16* __restrict__ W,
                            f16* __restrict__ dst, int mode) {
  __shared__ GemmSmem s;
  const int tid  = threadIdx.x;
  const int lane = tid & 31;
  const int wv   = tid >> 5;
  const int rb   = blockIdx.x >> 5;          // 32 row blocks of 128
  const int h    = (blockIdx.x >> 1) & 15;   // head
  const int g    = blockIdx.x & 1;           // RoPE column group
  const int r0   = rb * 128;
  const int e0   = h * DH + 32 * g;
  const int hl   = lane >> 4;

  v8f c[4] = {};
  gemm_mainloop<true>(s, X, W, r0, e0, wv, tid, lane, c);

  const int r0w    = r0 + 16 * wv;
  const int b_     = r0w / SEQ;
  const int t_base = (r0w & (SEQ - 1)) + 8 * hl;
  const int dbase  = 32 * g;                 // in-head column base (first half)

  if (mode < 2) {
    const float c2 = 0.20762050593046937f;   // log2(10000)/64
#pragma unroll
    for (int j = 0; j < 2; ++j) {            // pairs (j, j+2): dp and dp+64
      float inv = exp2f(-c2 * (float)(dbase + 16 * j + (lane & 15)));
#pragma unroll
      for (int v = 0; v < 8; ++v) {
        float ang = (float)(t_base + v) * inv;
        float sn, cs;
        sincosf(ang, &sn, &cs);
        float x1 = c[j][v], x2 = c[j + 2][v];
        c[j][v]     = x1 * cs - x2 * sn;
        c[j + 2][v] = x1 * sn + x2 * cs;
      }
    }
  }
  const float oscale = (mode == 0) ? 0.12752551286084332f : 1.0f; // log2e/sqrt(128)

  const size_t hb = (size_t)b_ * NH + h;
#pragma unroll
  for (int j = 0; j < 4; ++j) {
    int dp = dbase + 16 * j + ((j >= 2) ? 32 : 0) + (lane & 15);  // j>=2 -> +64 half
#pragma unroll
    for (int v = 0; v < 8; ++v) {
      int t = t_base + v;
      float val = c[j][v] * oscale;
      size_t idx = (mode == 2) ? (hb * DH + dp) * SEQ + t    // V: [b,h,d,t]
                               : (hb * SEQ + t) * DH + dp;   // Q,K: [b,h,t,d]
      dst[idx] = (f16)val;
    }
  }
}

// ---------------------------------------------------------------------------
// Flash attention: one wave per 16-query tile per (b,h); 32 keys per step.
// ---------------------------------------------------------------------------
__global__ void attn_kernel(const f16* __restrict__ Q, const f16* __restrict__ K,
                            const f16* __restrict__ V, f16* __restrict__ CTX) {
  __shared__ __align__(16) f16 lds[8][16 * 32];   // 1KB per wave

  const int lane = threadIdx.x & 31;
  const int wib  = threadIdx.x >> 5;
  const int w    = blockIdx.x * (blockDim.x >> 5) + wib;
  const int qt   = w & 127;
  const int h    = (w >> 7) & 15;
  const int b    = w >> 11;
  const int t0   = qt * 16;
  const int hl   = lane >> 4;

  const f16* qb = Q + ((size_t)b * NH + h) * SEQ * DH;
  const f16* kb = K + ((size_t)b * NH + h) * SEQ * DH;
  const f16* vb = V + ((size_t)b * NH + h) * DH * SEQ;

  v16h qa[4];
#pragma unroll
  for (int s = 0; s < 4; ++s) qa[s] = load_a_frag(qb, DH, t0, 32 * s, lane);

  v8f o[8] = {};
  float m[8], l[8];
#pragma unroll
  for (int v = 0; v < 8; ++v) { m[v] = -INFINITY; l[v] = 0.f; }

  const int kb_end = (t0 + 15) >> 5;
#pragma unroll 1
  for (int kblk = 0; kblk <= kb_end; ++kblk) {
    const int k0 = 32 * kblk;
    v8f s0 = {}, s1 = {};
#pragma unroll
    for (int s = 0; s < 4; ++s) {
      v16h kf0 = load_b_frag(kb, DH, k0,      32 * s, lane);
      v16h kf1 = load_b_frag(kb, DH, k0 + 16, 32 * s, lane);
      s0 = wmma_f16(qa[s], kf0, s0);
      s1 = wmma_f16(qa[s], kf1, s1);
    }
    if (k0 + 31 > t0) {                         // diagonal: causal mask
#pragma unroll
      for (int v = 0; v < 8; ++v) {
        int q   = t0 + v + 8 * hl;
        int key = k0 + (lane & 15);
        if (key > q)      s0[v] = -3.0e38f;
        if (key + 16 > q) s1[v] = -3.0e38f;
      }
    }
#pragma unroll
    for (int v = 0; v < 8; ++v) {               // online softmax (base-2)
      float t = fmaxf(s0[v], s1[v]);
      t = fmaxf(t, __shfl_xor(t, 1));
      t = fmaxf(t, __shfl_xor(t, 2));
      t = fmaxf(t, __shfl_xor(t, 4));
      t = fmaxf(t, __shfl_xor(t, 8));
      float mn    = fmaxf(m[v], t);
      float alpha = exp2f(m[v] - mn);
      m[v] = mn;
      s0[v] = exp2f(s0[v] - mn);
      s1[v] = exp2f(s1[v] - mn);
      float rs = s0[v] + s1[v];
      rs += __shfl_xor(rs, 1);
      rs += __shfl_xor(rs, 2);
      rs += __shfl_xor(rs, 4);
      rs += __shfl_xor(rs, 8);
      l[v] = l[v] * alpha + rs;
#pragma unroll
      for (int j = 0; j < 8; ++j) o[j][v] *= alpha;
    }
    // P: C-layout -> row-major 16x32 LDS -> A-fragment (cross-lane transpose)
    f16* pl = &lds[wib][0];
#pragma unroll
    for (int v = 0; v < 8; ++v) {
      int row = v + 8 * hl;
      pl[row * 32 + (lane & 15)]      = (f16)s0[v];
      pl[row * 32 + 16 + (lane & 15)] = (f16)s1[v];
    }
    asm volatile("s_wait_dscnt 0" ::: "memory");
    v16h pa;
    {
      const f16* p = pl + (lane & 15) * 32 + 8 * hl;
      pa = cat16(*(const v8h*)p, *(const v8h*)(p + 16));
    }
#pragma unroll
    for (int j = 0; j < 8; ++j) {               // O(16x128) += P(16x32)*V
      v16h vf = load_b_frag(vb, SEQ, 16 * j, k0, lane);
      o[j] = wmma_f16(pa, vf, o[j]);
    }
  }
#pragma unroll
  for (int v = 0; v < 8; ++v) l[v] = 1.0f / l[v];
#pragma unroll
  for (int j = 0; j < 8; ++j) {
    int dp = 16 * j + (lane & 15);
#pragma unroll
    for (int v = 0; v < 8; ++v) {
      int t = t0 + v + 8 * hl;
      CTX[((size_t)b * SEQ + t) * DM + h * DH + dp] = (f16)(o[j][v] * l[v]);
    }
  }
}

// ---------------------------------------------------------------------------
// Output projection: out[r,e] = sum_d ctx[r,d] * Wo[e,d]; fp32 out; same
// async-LDS GEMM core with contiguous 64-column tiles.
// ---------------------------------------------------------------------------
__global__ void oproj_kernel(const f16* __restrict__ CTX, const f16* __restrict__ Wo,
                             float* __restrict__ out) {
  __shared__ GemmSmem s;
  const int tid  = threadIdx.x;
  const int lane = tid & 31;
  const int wv   = tid >> 5;
  const int rb   = blockIdx.x >> 5;          // 32 row blocks of 128
  const int eb   = blockIdx.x & 31;          // 32 col blocks of 64
  const int r0   = rb * 128;
  const int e0   = eb * 64;
  const int hl   = lane >> 4;

  v8f c[4] = {};
  gemm_mainloop<false>(s, CTX, Wo, r0, e0, wv, tid, lane, c);

  const int r0w = r0 + 16 * wv;
#pragma unroll
  for (int j = 0; j < 4; ++j) {
    int e = e0 + 16 * j + (lane & 15);
#pragma unroll
    for (int v = 0; v < 8; ++v)
      out[(size_t)(r0w + v + 8 * hl) * DM + e] = c[j][v];
  }
}

// ---------------------------------------------------------------------------
extern "C" void kernel_launch(void* const* d_in, const int* in_sizes, int n_in,
                              void* d_out, int out_size, void* d_ws, size_t ws_size,
                              hipStream_t stream) {
  const float* x  = (const float*)d_in[0];
  const float* Wq = (const float*)d_in[1];
  const float* Wk = (const float*)d_in[2];
  const float* Wv = (const float*)d_in[3];
  const float* Wo = (const float*)d_in[4];
  float* out = (float*)d_out;

  char* ws = (char*)d_ws;
  size_t off = 0;
  auto carve = [&](size_t bytes) -> f16* {
    f16* p = (f16*)(ws + off);
    off += (bytes + 255) & ~(size_t)255;
    return p;
  };
  const size_t xw_bytes = (size_t)ROWS * DM * sizeof(f16);
  const size_t w_bytes  = (size_t)DM * DM * sizeof(f16);
  const size_t h_bytes  = (size_t)BATCH * NH * SEQ * DH * sizeof(f16);

  f16* Xh  = carve(xw_bytes);
  f16* Wqh = carve(w_bytes);
  f16* Wkh = carve(w_bytes);
  f16* Wvh = carve(w_bytes);
  f16* Woh = carve(w_bytes);
  f16* Qb  = carve(h_bytes);
  f16* Kb  = carve(h_bytes);
  f16* Vb  = carve(h_bytes);
  f16* CTX = carve(xw_bytes);   // total ~118 MB

  cvt_f32_f16_kernel<<<1024, 256, 0, stream>>>(x,  Xh,  ROWS * DM);
  cvt_f32_f16_kernel<<<512,  256, 0, stream>>>(Wq, Wqh, DM * DM);
  cvt_f32_f16_kernel<<<512,  256, 0, stream>>>(Wk, Wkh, DM * DM);
  cvt_f32_f16_kernel<<<512,  256, 0, stream>>>(Wv, Wvh, DM * DM);
  cvt_f32_f16_kernel<<<512,  256, 0, stream>>>(Wo, Woh, DM * DM);

  // 1024 workgroups x 8 waves: 128x64 tiles (proj: head quarters {g,g+2})
  proj_kernel<<<1024, 256, 0, stream>>>(Xh, Wqh, Qb, 0);
  proj_kernel<<<1024, 256, 0, stream>>>(Xh, Wkh, Kb, 1);
  proj_kernel<<<1024, 256, 0, stream>>>(Xh, Wvh, Vb, 2);

  attn_kernel<<<512, 256, 0, stream>>>(Qb, Kb, Vb, CTX);

  oproj_kernel<<<1024, 256, 0, stream>>>(CTX, Woh, out);
}
